// CrossKGAttention_25082609008835
// MI455X (gfx1250) — compile-verified, matrix operand order checked by reading, not run on hardware
//
#include <hip/hip_runtime.h>
#include <hip/hip_bf16.h>

#define HIDDEN   256
#define HEADS    4
#define HEAD_DIM 64
#define SCALE    0.125f   // 64^-0.5
#define QB       32       // queries per block (2 x 16 tiles per wave)

typedef __attribute__((ext_vector_type(16))) _Float16 v16h;
typedef __attribute__((ext_vector_type(8)))  float    v8f;

// ---------------- WMMA helpers (CDNA5 16x16x32 f16, fp32 accum) ----------------

__device__ __forceinline__ v8f zero8() {
  v8f z;
#pragma unroll
  for (int i = 0; i < 8; ++i) z[i] = 0.f;
  return z;
}

// 16x32 f16 fragment load from a row-major [row][k] tile (A layout; B layout is
// the same pattern over an [n][k] tile). ISA 7.12.2: lanes 0-15 rows 0-15;
// elements e<8 -> k=e+koff, e>=8 -> k=e+8+koff, koff=8 for lanes 16-31.
__device__ __forceinline__ v16h load_frag_row(const _Float16* base, int ld, int row) {
  const int lane = threadIdx.x & 31;
  const int koff = (lane >> 4) << 3;
  const _Float16* p = base + (size_t)row * ld + koff;
  v16h f;
#pragma unroll
  for (int e = 0; e < 8; ++e) f[e] = p[e];
#pragma unroll
  for (int e = 0; e < 8; ++e) f[8 + e] = p[16 + e];
  return f;
}

__device__ __forceinline__ v16h load_frag(const _Float16* base, int ld) {
  return load_frag_row(base, ld, threadIdx.x & 15);
}

__device__ __forceinline__ v8f wmma_f16(v16h a, v16h b, v8f c) {
  return __builtin_amdgcn_wmma_f32_16x16x32_f16(false, a, false, b, (short)0, c,
                                                false, false);
}

// ---------------- async global->LDS copy (CDNA5, ASYNCcnt) ----------------

__device__ __forceinline__ uint32_t lds_u32(const void* p) {
  // generic shared-aperture address: low 32 bits are the LDS byte address
  return (uint32_t)(uintptr_t)p;
}

__device__ __forceinline__ void async_copy16(uint32_t lds, const void* g) {
  asm volatile("global_load_async_to_lds_b128 %0, %1, off"
               :: "v"(lds), "v"((uint64_t)(uintptr_t)g) : "memory");
}

__device__ __forceinline__ void wait_async0() {
  asm volatile("s_wait_asynccnt 0" ::: "memory");
}

// ---------------- prep kernels ----------------

__global__ void k_cvt_f32_f16(const float* __restrict__ src,
                              _Float16* __restrict__ dst, int n) {
  int i = blockIdx.x * blockDim.x + threadIdx.x;
  if (i < n) dst[i] = (_Float16)src[i];
}

// Fold head-mean into Wk: Wkbar[d][k] = mean_h Wk[h*64+d][k]; bkbar likewise.
__global__ void k_make_wkbar(const float* __restrict__ Wk,
                             const float* __restrict__ bk,
                             _Float16* __restrict__ wkb,
                             float* __restrict__ bkb) {
  int i = blockIdx.x * blockDim.x + threadIdx.x;
  if (i < HEAD_DIM * HIDDEN) {
    int d = i >> 8, k = i & 255;
    float s = 0.25f * (Wk[(size_t)(0 * HEAD_DIM + d) * HIDDEN + k] +
                       Wk[(size_t)(1 * HEAD_DIM + d) * HIDDEN + k] +
                       Wk[(size_t)(2 * HEAD_DIM + d) * HIDDEN + k] +
                       Wk[(size_t)(3 * HEAD_DIM + d) * HIDDEN + k]);
    wkb[i] = (_Float16)s;
  }
  if (i < HEAD_DIM)
    bkb[i] = 0.25f * (bk[i] + bk[HEAD_DIM + i] + bk[2 * HEAD_DIM + i] +
                      bk[3 * HEAD_DIM + i]);
}

// ---------------- WMMA GEMM: C = A[M,K] @ Bt[N,K]^T + bias ----------------
// One wave per 16x16 output tile; 4 waves per block.

__global__ void __launch_bounds__(128)
k_gemm_bias_h(const _Float16* __restrict__ A, const _Float16* __restrict__ Bt,
              const float* __restrict__ bias, _Float16* __restrict__ C,
              int M, int N, int K) {
  const int ntilesN = N >> 4;
  const int tiles = (M >> 4) * ntilesN;
  int t = blockIdx.x * 4 + ((int)threadIdx.x >> 5);
  if (t >= tiles) return;
  int mt = t / ntilesN, nt = t % ntilesN;
  const _Float16* a = A + (size_t)mt * 16 * K;
  const _Float16* b = Bt + (size_t)nt * 16 * K;
  v8f c = zero8();
  for (int k0 = 0; k0 < K; k0 += 32)
    c = wmma_f16(load_frag(a + k0, K), load_frag(b + k0, K), c);
  const int lane = threadIdx.x & 31;
  const int n = lane & 15, mb = (lane >> 4) << 3;
  float bn = bias[nt * 16 + n];
#pragma unroll
  for (int r = 0; r < 8; ++r)
    C[(size_t)(mt * 16 + mb + r) * N + nt * 16 + n] = (_Float16)(c[r] + bn);
}

// Same, but stores the TRANSPOSED result: Ct[N][M] (for V^T used in attention).
__global__ void __launch_bounds__(128)
k_gemm_bias_ht(const _Float16* __restrict__ A, const _Float16* __restrict__ Bt,
               const float* __restrict__ bias, _Float16* __restrict__ Ct,
               int M, int N, int K) {
  const int ntilesN = N >> 4;
  const int tiles = (M >> 4) * ntilesN;
  int t = blockIdx.x * 4 + ((int)threadIdx.x >> 5);
  if (t >= tiles) return;
  int mt = t / ntilesN, nt = t % ntilesN;
  const _Float16* a = A + (size_t)mt * 16 * K;
  const _Float16* b = Bt + (size_t)nt * 16 * K;
  v8f c = zero8();
  for (int k0 = 0; k0 < K; k0 += 32)
    c = wmma_f16(load_frag(a + k0, K), load_frag(b + k0, K), c);
  const int lane = threadIdx.x & 31;
  const int n = lane & 15, mb = (lane >> 4) << 3;
  float bn = bias[nt * 16 + n];
#pragma unroll
  for (int r = 0; r < 8; ++r)
    Ct[(size_t)(nt * 16 + n) * M + mt * 16 + mb + r] = (_Float16)(c[r] + bn);
}

// fp32 output with residual add (final Wo projection).
__global__ void __launch_bounds__(128)
k_gemm_bias_res_f(const _Float16* __restrict__ A, const _Float16* __restrict__ Bt,
                  const float* __restrict__ bias, const float* __restrict__ res,
                  float* __restrict__ out, int M, int N, int K) {
  const int ntilesN = N >> 4;
  const int tiles = (M >> 4) * ntilesN;
  int t = blockIdx.x * 4 + ((int)threadIdx.x >> 5);
  if (t >= tiles) return;
  int mt = t / ntilesN, nt = t % ntilesN;
  const _Float16* a = A + (size_t)mt * 16 * K;
  const _Float16* b = Bt + (size_t)nt * 16 * K;
  v8f c = zero8();
  for (int k0 = 0; k0 < K; k0 += 32)
    c = wmma_f16(load_frag(a + k0, K), load_frag(b + k0, K), c);
  const int lane = threadIdx.x & 31;
  const int n = lane & 15, mb = (lane >> 4) << 3;
  float bn = bias[nt * 16 + n];
#pragma unroll
  for (int r = 0; r < 8; ++r) {
    size_t idx = (size_t)(mt * 16 + mb + r) * N + nt * 16 + n;
    out[idx] = res[idx] + c[r] + bn;
  }
}

// ---------------- fused masked cross-attention (flash-style) ----------------
// 128 threads = 4 waves; wave = head; 32 queries (2 q-tiles) per block.
// kv streamed in tiles of 32 with async global->LDS double buffering.
// Out-of-range kv handled by address clamping + -inf masking (P weight = 0,
// so clamped garbage never contributes). Requires Nk % 8 == 0 for V chunks
// (6000 ok) and Nq % 4 == 0 for transposed mask chunks (6000 ok).

__device__ __forceinline__ void issue_tile_async(
    int tid, int kv0, int q0, int Nq, int Nk,
    const _Float16* __restrict__ Kb, const _Float16* __restrict__ Vt,
    const float* __restrict__ align, int stride, int transposed,
    _Float16* s_kbar, _Float16* s_vt, float* s_align) {
  // Kbar tile [kv][64], 256 x 16B chunks
  for (int j = tid; j < 256; j += 128) {
    int kv = j >> 3, d8 = (j & 7) << 3;
    int g = kv0 + kv; if (g > Nk - 1) g = Nk - 1;
    async_copy16(lds_u32(s_kbar + kv * HEAD_DIM + d8),
                 Kb + (size_t)g * HEAD_DIM + d8);
  }
  // V^T tile [c=256][kv=32], 1024 x 16B chunks (contiguous both sides)
  for (int j = tid; j < 1024; j += 128) {
    int c = j >> 2, kv8 = (j & 3) << 3;
    int g = kv0 + kv8; if (g > Nk - 8) g = Nk - 8;
    async_copy16(lds_u32(s_vt + c * 32 + kv8),
                 Vt + (size_t)c * Nk + g);
  }
  // mask tile: [q][kv] normal / [kv][q] transposed, 256 x 16B chunks
  if (!transposed) {
    for (int j = tid; j < 256; j += 128) {
      int q = j >> 3, kv4 = (j & 7) << 2;
      int gq = q0 + q;    if (gq > Nq - 1) gq = Nq - 1;
      int gk = kv0 + kv4; if (gk > Nk - 4) gk = Nk - 4;
      async_copy16(lds_u32(s_align + q * 32 + kv4),
                   align + (size_t)gq * stride + gk);
    }
  } else {
    for (int j = tid; j < 256; j += 128) {
      int kv = j >> 3, q4 = (j & 7) << 2;
      int gk = kv0 + kv; if (gk > Nk - 1) gk = Nk - 1;
      int gq = q0 + q4;  if (gq > Nq - 4) gq = Nq - 4;
      async_copy16(lds_u32(s_align + kv * 32 + q4),
                   align + (size_t)gk * stride + gq);
    }
  }
}

__global__ void __launch_bounds__(128)
k_attn(const _Float16* __restrict__ Q, const _Float16* __restrict__ Kb,
       const _Float16* __restrict__ Vt, const float* __restrict__ align,
       int alignStride, int transposed, int Nq, int Nk,
       _Float16* __restrict__ O) {
  __shared__ _Float16 s_kbar[2][32 * HEAD_DIM];   // [kv][d]
  __shared__ _Float16 s_vt[2][HIDDEN * 32];       // [h*64+d][kv]
  __shared__ float    s_align[2][32 * 32];        // [q][kv] or [kv][q]
  __shared__ _Float16 s_p[HEADS][16 * 32];        // per-head P tile

  const int tid  = threadIdx.x;
  const int head = tid >> 5;
  const int lane = tid & 31;
  const int q0 = blockIdx.x * QB;
  const int n  = lane & 15;
  const int mb = (lane >> 4) << 3;

  // Persistent Q fragments, 2 q-tiles x (K=0..31, K=32..63), rows clamped.
  v16h qa[2][2];
#pragma unroll
  for (int qt = 0; qt < 2; ++qt) {
    int qr = q0 + qt * 16 + (lane & 15);
    if (qr > Nq - 1) qr = Nq - 1;
    qa[qt][0] = load_frag_row(Q + head * HEAD_DIM + 0,  HIDDEN, qr);
    qa[qt][1] = load_frag_row(Q + head * HEAD_DIM + 32, HIDDEN, qr);
  }

  v8f o[2][4];
  float mrun[2][8], lrun[2][8];
#pragma unroll
  for (int qt = 0; qt < 2; ++qt) {
#pragma unroll
    for (int dt = 0; dt < 4; ++dt) o[qt][dt] = zero8();
#pragma unroll
    for (int r = 0; r < 8; ++r) { mrun[qt][r] = -3.0e38f; lrun[qt][r] = 0.f; }
  }

  const int nkt = (Nk + 31) >> 5;
  issue_tile_async(tid, 0, q0, Nq, Nk, Kb, Vt, align, alignStride, transposed,
                   s_kbar[0], s_vt[0], s_align[0]);
  wait_async0();
  __syncthreads();

  for (int kt = 0; kt < nkt; ++kt) {
    const int kv0 = kt << 5;
    const int buf = kt & 1;
    if (kt + 1 < nkt)
      issue_tile_async(tid, (kt + 1) << 5, q0, Nq, Nk, Kb, Vt, align,
                       alignStride, transposed,
                       s_kbar[buf ^ 1], s_vt[buf ^ 1], s_align[buf ^ 1]);

    const _Float16* kbb = s_kbar[buf];
    const _Float16* vtb = s_vt[buf];
    const float*    alb = s_align[buf];
    const bool vk0 = (kv0 + n) < Nk;
    const bool vk1 = (kv0 + 16 + n) < Nk;

#pragma unroll
    for (int qt = 0; qt < 2; ++qt) {
      // scores: 2 kv sub-tiles x (K=64 -> 2 WMMAs)
      v8f c0 = zero8(), c1 = zero8();
      c0 = wmma_f16(qa[qt][0], load_frag(kbb + 0,            HEAD_DIM), c0);
      c0 = wmma_f16(qa[qt][1], load_frag(kbb + 32,           HEAD_DIM), c0);
      c1 = wmma_f16(qa[qt][0], load_frag(kbb + 16 * 64 + 0,  HEAD_DIM), c1);
      c1 = wmma_f16(qa[qt][1], load_frag(kbb + 16 * 64 + 32, HEAD_DIM), c1);

      float s0[8], s1[8];
#pragma unroll
      for (int r = 0; r < 8; ++r) {
        int mq = qt * 16 + mb + r;
        float a0 = transposed ? alb[n * 32 + mq]        : alb[mq * 32 + n];
        float a1 = transposed ? alb[(n + 16) * 32 + mq] : alb[mq * 32 + n + 16];
        s0[r] = vk0 ? c0[r] * (SCALE * a0) : -3.0e38f;
        s1[r] = vk1 ? c1[r] * (SCALE * a1) : -3.0e38f;
      }

      // online softmax (row reductions across 16-lane column groups)
      float pr0[8], pr1[8];
#pragma unroll
      for (int r = 0; r < 8; ++r) {
        float t = fmaxf(s0[r], s1[r]);
#pragma unroll
        for (int off = 8; off; off >>= 1) t = fmaxf(t, __shfl_xor(t, off, 32));
        float nm = fmaxf(mrun[qt][r], t);
        float rs = __expf(mrun[qt][r] - nm);
        mrun[qt][r] = nm;
        float p0 = __expf(s0[r] - nm);
        float p1 = __expf(s1[r] - nm);
        pr0[r] = p0; pr1[r] = p1;
        float sum = p0 + p1;
#pragma unroll
        for (int off = 8; off; off >>= 1) sum += __shfl_xor(sum, off, 32);
        lrun[qt][r] = lrun[qt][r] * rs + sum;
#pragma unroll
        for (int dt = 0; dt < 4; ++dt) o[qt][dt][r] *= rs;
      }

      // P tile to LDS (A-fragment order), then P @ V_head
      _Float16* ph = s_p[head];
#pragma unroll
      for (int r = 0; r < 8; ++r) {
        ph[(mb + r) * 32 + n]      = (_Float16)pr0[r];
        ph[(mb + r) * 32 + n + 16] = (_Float16)pr1[r];
      }
      v16h pa = load_frag(ph, 32);
#pragma unroll
      for (int dt = 0; dt < 4; ++dt)
        o[qt][dt] = wmma_f16(pa,
                             load_frag(vtb + (head * HEAD_DIM + dt * 16) * 32, 32),
                             o[qt][dt]);
    }

    wait_async0();
    __syncthreads();
  }

#pragma unroll
  for (int qt = 0; qt < 2; ++qt)
#pragma unroll
    for (int r = 0; r < 8; ++r) {
      int gq = q0 + qt * 16 + mb + r;
      if (gq < Nq) {
        float inv = 1.0f / lrun[qt][r];
        size_t row = (size_t)gq * HIDDEN + head * HEAD_DIM + n;
        O[row + 0]  = (_Float16)(o[qt][0][r] * inv);
        O[row + 16] = (_Float16)(o[qt][1][r] * inv);
        O[row + 32] = (_Float16)(o[qt][2][r] * inv);
        O[row + 48] = (_Float16)(o[qt][3][r] * inv);
      }
    }
}

// ---------------- host orchestration ----------------

static inline char* bump_alloc(char*& p, size_t bytes) {
  char* r = p;
  p += (bytes + 255) & ~(size_t)255;
  return r;
}

extern "C" void kernel_launch(void* const* d_in, const int* in_sizes, int n_in,
                              void* d_out, int out_size, void* d_ws, size_t ws_size,
                              hipStream_t stream) {
  const float* kg1  = (const float*)d_in[0];
  const float* kg2  = (const float*)d_in[1];
  const float* algn = (const float*)d_in[2];
  const float* Wq = (const float*)d_in[3];  const float* bq = (const float*)d_in[4];
  const float* Wk = (const float*)d_in[5];  const float* bk = (const float*)d_in[6];
  const float* Wv = (const float*)d_in[7];  const float* bv = (const float*)d_in[8];
  const float* Wo = (const float*)d_in[9];  const float* bo = (const float*)d_in[10];
  const int N1 = in_sizes[0] / HIDDEN;
  const int N2 = in_sizes[1] / HIDDEN;

  char* p = (char*)d_ws;
  _Float16* e1h  = (_Float16*)bump_alloc(p, (size_t)N1 * HIDDEN * 2);
  _Float16* e2h  = (_Float16*)bump_alloc(p, (size_t)N2 * HIDDEN * 2);
  _Float16* wqh  = (_Float16*)bump_alloc(p, (size_t)HIDDEN * HIDDEN * 2);
  _Float16* wvh  = (_Float16*)bump_alloc(p, (size_t)HIDDEN * HIDDEN * 2);
  _Float16* woh  = (_Float16*)bump_alloc(p, (size_t)HIDDEN * HIDDEN * 2);
  _Float16* wkb  = (_Float16*)bump_alloc(p, (size_t)HEAD_DIM * HIDDEN * 2);
  float*    bkb  = (float*)   bump_alloc(p, (size_t)HEAD_DIM * 4);
  _Float16* Q1   = (_Float16*)bump_alloc(p, (size_t)N1 * HIDDEN * 2);
  _Float16* Q2   = (_Float16*)bump_alloc(p, (size_t)N2 * HIDDEN * 2);
  _Float16* V1t  = (_Float16*)bump_alloc(p, (size_t)N1 * HIDDEN * 2);
  _Float16* V2t  = (_Float16*)bump_alloc(p, (size_t)N2 * HIDDEN * 2);
  _Float16* Kb1  = (_Float16*)bump_alloc(p, (size_t)N1 * HEAD_DIM * 2);
  _Float16* Kb2  = (_Float16*)bump_alloc(p, (size_t)N2 * HEAD_DIM * 2);
  _Float16* at1  = (_Float16*)bump_alloc(p, (size_t)N1 * HIDDEN * 2);
  _Float16* at2  = (_Float16*)bump_alloc(p, (size_t)N2 * HIDDEN * 2);

  auto cvt = [&](const float* s, _Float16* d, int n) {
    k_cvt_f32_f16<<<(n + 255) / 256, 256, 0, stream>>>(s, d, n);
  };
  cvt(kg1, e1h, N1 * HIDDEN);
  cvt(kg2, e2h, N2 * HIDDEN);
  cvt(Wq, wqh, HIDDEN * HIDDEN);
  cvt(Wv, wvh, HIDDEN * HIDDEN);
  cvt(Wo, woh, HIDDEN * HIDDEN);
  k_make_wkbar<<<(HEAD_DIM * HIDDEN + 255) / 256, 256, 0, stream>>>(Wk, bk, wkb, bkb);

  auto gemmH = [&](const _Float16* A, const _Float16* Bt, const float* bias,
                   _Float16* C, int M, int N, int K) {
    int tiles = (M / 16) * (N / 16);
    k_gemm_bias_h<<<(tiles + 3) / 4, 128, 0, stream>>>(A, Bt, bias, C, M, N, K);
  };
  auto gemmHT = [&](const _Float16* A, const _Float16* Bt, const float* bias,
                    _Float16* Ct, int M, int N, int K) {
    int tiles = (M / 16) * (N / 16);
    k_gemm_bias_ht<<<(tiles + 3) / 4, 128, 0, stream>>>(A, Bt, bias, Ct, M, N, K);
  };
  gemmH (e1h, wqh, bq,  Q1,  N1, HIDDEN,   HIDDEN);
  gemmH (e2h, wqh, bq,  Q2,  N2, HIDDEN,   HIDDEN);
  gemmHT(e1h, wvh, bv,  V1t, N1, HIDDEN,   HIDDEN);   // stores V^T [256][N1]
  gemmHT(e2h, wvh, bv,  V2t, N2, HIDDEN,   HIDDEN);   // stores V^T [256][N2]
  gemmH (e1h, wkb, bkb, Kb1, N1, HEAD_DIM, HIDDEN);
  gemmH (e2h, wkb, bkb, Kb2, N2, HEAD_DIM, HIDDEN);

  // dir1: queries = kg1, kv = kg2, mask = align[q][kv]
  k_attn<<<(N1 + QB - 1) / QB, 128, 0, stream>>>(Q1, Kb2, V2t, algn, N2, 0,
                                                 N1, N2, at1);
  // dir2: queries = kg2, kv = kg1, mask = align[kv][q] (transposed)
  k_attn<<<(N2 + QB - 1) / QB, 128, 0, stream>>>(Q2, Kb1, V1t, algn, N2, 1,
                                                 N2, N1, at2);

  float* out = (float*)d_out;
  int t1 = (N1 / 16) * (HIDDEN / 16);
  int t2 = (N2 / 16) * (HIDDEN / 16);
  k_gemm_bias_res_f<<<(t1 + 3) / 4, 128, 0, stream>>>(at1, woh, bo, kg1, out,
                                                      N1, HIDDEN, HIDDEN);
  k_gemm_bias_res_f<<<(t2 + 3) / 4, 128, 0, stream>>>(at2, woh, bo, kg2,
                                                      out + (size_t)N1 * HIDDEN,
                                                      N2, HIDDEN, HIDDEN);
}
